// SpectralAutoencoder_24352464570164
// MI455X (gfx1250) — compile-verified, matrix operand order checked by reading, not run on hardware
//
#include <hip/hip_runtime.h>
#include <hip/hip_bf16.h>
#include <math.h>

#define D_MODEL 768
#define D_LAT   384
#define NFREQ   385      // D_MODEL/2 + 1
#define ROWS    32768    // 8 * 4096

typedef __attribute__((ext_vector_type(16))) _Float16 v16h;
typedef __attribute__((ext_vector_type(8)))  _Float16 v8h;
typedef __attribute__((ext_vector_type(8)))  float    v8f;
typedef __attribute__((ext_vector_type(4)))  float    v4f;

// ---------------------------------------------------------------------------
// WMMA fragment loaders (wave32, 16x16x32 f16).
// A 16x32 f16 layout: lane holds row m=lane&15; halves e=0..7 -> K = 8*(lane>>4)+e,
// halves e=8..15 -> K = 16 + 8*(lane>>4) + (e-8). Two contiguous 16B chunks.
// B stored transposed [n][k] so the same pattern applies with n = lane&15.
// ---------------------------------------------------------------------------
__device__ __forceinline__ v16h load_frag_a_f32(const float* __restrict__ A, int lda,
                                                int row0, int k0, int lane) {
  int m = lane & 15, kh = lane >> 4;
  const float* p = A + (size_t)(row0 + m) * lda + (k0 + 8 * kh);
  v4f c0 = *(const v4f*)(p + 0);
  v4f c1 = *(const v4f*)(p + 4);
  v4f c2 = *(const v4f*)(p + 16);
  v4f c3 = *(const v4f*)(p + 20);
  v16h r;
#pragma unroll
  for (int e = 0; e < 4; ++e) {
    r[e]      = (_Float16)c0[e];
    r[e + 4]  = (_Float16)c1[e];
    r[e + 8]  = (_Float16)c2[e];
    r[e + 12] = (_Float16)c3[e];
  }
  return r;
}

__device__ __forceinline__ v16h load_frag_f16(const _Float16* __restrict__ B, int ldb,
                                              int row0, int k0, int lane) {
  int m = lane & 15, kh = lane >> 4;
  const _Float16* p = B + (size_t)(row0 + m) * ldb + (k0 + 8 * kh);
  v8h c0 = *(const v8h*)(p);
  v8h c1 = *(const v8h*)(p + 16);
  v16h r;
#pragma unroll
  for (int e = 0; e < 8; ++e) { r[e] = c0[e]; r[e + 8] = c1[e]; }
  return r;
}

// ---------------------------------------------------------------------------
// Precompute kernels (all tiny).
// ---------------------------------------------------------------------------

// k[t] = (1/n)( kr[0] + 2*sum_{f=1..n/2-1}(kr[f]cos(2pi f t/n) - ki[f]sin(2pi f t/n))
//               + kr[n/2]*(-1)^t )   == irfft(kr + i*ki, n)
__global__ void k_irfft768(const float* __restrict__ kr, const float* __restrict__ ki,
                           float* __restrict__ kout) {
  int t = threadIdx.x;
  float s = kr[0];
  for (int f = 1; f < NFREQ - 1; ++f) {
    int ph = (f * t) % D_MODEL;  // exact integer phase reduction
    float th = (float)ph * (2.0f * 3.14159265358979323846f / (float)D_MODEL);
    s += 2.0f * (kr[f] * cosf(th) - ki[f] * sinf(th));
  }
  s += kr[NFREQ - 1] * ((t & 1) ? -1.0f : 1.0f);
  kout[t] = s * (1.0f / (float)D_MODEL);
}

// Circulant, transposed for B-fragment loads: Bt[t][s] = k[(t-s) mod n]  (fp16)
__global__ void k_build_Bt(const float* __restrict__ kvec, _Float16* __restrict__ Bt) {
  int idx = blockIdx.x * blockDim.x + threadIdx.x;
  if (idx >= D_MODEL * D_MODEL) return;
  int t = idx / D_MODEL, s = idx % D_MODEL;
  Bt[idx] = (_Float16)kvec[(t - s + D_MODEL) % D_MODEL];
}

// Encoder quaternion matrix, output-major [o][i] (o<384, i<768), fp16.
__global__ void k_build_Menc(const float* __restrict__ Ww, const float* __restrict__ Wx,
                             const float* __restrict__ Wy, const float* __restrict__ Wz,
                             _Float16* __restrict__ M) {
  int idx = blockIdx.x * blockDim.x + threadIdx.x;
  if (idx >= D_LAT * D_MODEL) return;
  int o = idx / D_MODEL, i = idx % D_MODEL;
  int co = o / 96, r = o % 96, ci = i / 192, q = i % 192;
  const int   sel[4][4] = {{0,1,2,3},{1,0,3,2},{2,3,0,1},{3,2,1,0}};
  const float sg[4][4]  = {{1,-1,-1,-1},{1,1,-1,1},{1,1,1,-1},{1,-1,1,1}};
  const float* Ws[4] = {Ww, Wx, Wy, Wz};
  M[idx] = (_Float16)(sg[co][ci] * Ws[sel[co][ci]][r * 192 + q]);
}

// Decoder quaternion matrix, output-major [s][i] (s<768, i<384), fp32 (feeds M2 GEMM).
__global__ void k_build_Mdec(const float* __restrict__ Ww, const float* __restrict__ Wx,
                             const float* __restrict__ Wy, const float* __restrict__ Wz,
                             float* __restrict__ M) {
  int idx = blockIdx.x * blockDim.x + threadIdx.x;
  if (idx >= D_MODEL * D_LAT) return;
  int s = idx / D_LAT, i = idx % D_LAT;
  int co = s / 192, r = s % 192, ci = i / 96, q = i % 96;
  const int   sel[4][4] = {{0,1,2,3},{1,0,3,2},{2,3,0,1},{3,2,1,0}};
  const float sg[4][4]  = {{1,-1,-1,-1},{1,1,-1,1},{1,1,1,-1},{1,-1,1,1}};
  const float* Ws[4] = {Ww, Wx, Wy, Wz};
  M[idx] = sg[co][ci] * Ws[sel[co][ci]][r * 96 + q];
}

// Fold decoder filter into decoder weights:
// M2t[t][i] = sum_s Mdec_t[s][i] * kdec[(t-s) mod n]   (fp16, layout [t][i], ld=384)
// b2[t]    = sum_s dec_b[s]      * kdec[(t-s) mod n]
__global__ void k_build_M2(const float* __restrict__ Mdec, const float* __restrict__ kdec,
                           const float* __restrict__ dec_b,
                           _Float16* __restrict__ M2, float* __restrict__ b2) {
  int idx = blockIdx.x * blockDim.x + threadIdx.x;
  if (idx >= D_MODEL * D_LAT) return;
  int t = idx / D_LAT, i = idx % D_LAT;
  float s = 0.f;
  for (int u = 0; u < D_MODEL; ++u)
    s += Mdec[u * D_LAT + i] * kdec[(t - u + D_MODEL) % D_MODEL];
  M2[idx] = (_Float16)s;
  if (i == 0) {
    float bb = 0.f;
    for (int u = 0; u < D_MODEL; ++u)
      bb += dec_b[u] * kdec[(t - u + D_MODEL) % D_MODEL];
    b2[t] = bb;
  }
}

__global__ void k_zero(float* p) { if (threadIdx.x == 0 && blockIdx.x == 0) *p = 0.f; }

__global__ void k_final(const float* __restrict__ acc, float* __restrict__ out) {
  if (threadIdx.x == 0 && blockIdx.x == 0)
    out[0] = acc[0] * (1.0f / ((float)ROWS * (float)D_MODEL));
}

// ---------------------------------------------------------------------------
// GEMM 1: H = gelu(X (32768x768 f32) @ Cenc (768x768))  -> H f16
// Wave tile: 16(M) x 128(N); 6 N-tiles across 768.
// ---------------------------------------------------------------------------
__global__ __launch_bounds__(256) void k_gemm_filter_gelu(
    const float* __restrict__ X, const _Float16* __restrict__ Bt,
    _Float16* __restrict__ H) {
  int wid  = blockIdx.x * 8 + (threadIdx.x >> 5);
  int lane = threadIdx.x & 31;
  int mtile = wid / 6, ntile = wid % 6;
  int row0 = mtile << 4, col0 = ntile << 7;
  v8f acc[8] = {};
  for (int k0 = 0; k0 < D_MODEL; k0 += 32) {
    v16h a = load_frag_a_f32(X, D_MODEL, row0, k0, lane);
#pragma unroll
    for (int s = 0; s < 8; ++s) {
      v16h b = load_frag_f16(Bt, D_MODEL, col0 + s * 16, k0, lane);
      acc[s] = __builtin_amdgcn_wmma_f32_16x16x32_f16(false, a, false, b,
                                                      (short)0, acc[s], false, false);
    }
  }
  int n = lane & 15, mh = lane >> 4;
#pragma unroll
  for (int s = 0; s < 8; ++s) {
#pragma unroll
    for (int v = 0; v < 8; ++v) {
      float x = acc[s][v];
      float g = 0.5f * x * (1.0f + erff(x * 0.70710678118f));  // exact GELU
      H[(size_t)(row0 + v + 8 * mh) * D_MODEL + (col0 + s * 16 + n)] = (_Float16)g;
    }
  }
}

// ---------------------------------------------------------------------------
// GEMM 2: Z = H (32768x768 f16) @ Menc (768x384) + enc_b
// Writes Z fp32 (output 0) and Zh fp16 (feeds GEMM 3). 3 N-tiles across 384.
// ---------------------------------------------------------------------------
__global__ __launch_bounds__(256) void k_gemm_enc_quat(
    const _Float16* __restrict__ H, const _Float16* __restrict__ Mt,
    const float* __restrict__ bias, float* __restrict__ Zout,
    _Float16* __restrict__ Zh) {
  int wid  = blockIdx.x * 8 + (threadIdx.x >> 5);
  int lane = threadIdx.x & 31;
  int mtile = wid / 3, ntile = wid % 3;
  int row0 = mtile << 4, col0 = ntile << 7;
  v8f acc[8] = {};
  for (int k0 = 0; k0 < D_MODEL; k0 += 32) {
    v16h a = load_frag_f16(H, D_MODEL, row0, k0, lane);
#pragma unroll
    for (int s = 0; s < 8; ++s) {
      v16h b = load_frag_f16(Mt, D_MODEL, col0 + s * 16, k0, lane);
      acc[s] = __builtin_amdgcn_wmma_f32_16x16x32_f16(false, a, false, b,
                                                      (short)0, acc[s], false, false);
    }
  }
  int n = lane & 15, mh = lane >> 4;
#pragma unroll
  for (int s = 0; s < 8; ++s) {
    int col = col0 + s * 16 + n;
    float bv = bias[col];
#pragma unroll
    for (int v = 0; v < 8; ++v) {
      int row = row0 + v + 8 * mh;
      float z = acc[s][v] + bv;
      Zout[(size_t)row * D_LAT + col] = z;
      Zh[(size_t)row * D_LAT + col]   = (_Float16)z;
    }
  }
}

// ---------------------------------------------------------------------------
// GEMM 3 + loss: recon = Zh (32768x384 f16) @ M2 (384x768) + b2 ;
// accumulate sum((recon - X)^2) -> lossAcc. recon never hits memory.
// ---------------------------------------------------------------------------
__global__ __launch_bounds__(256) void k_gemm_dec_loss(
    const _Float16* __restrict__ Zh, const _Float16* __restrict__ M2t,
    const float* __restrict__ b2, const float* __restrict__ X,
    float* __restrict__ lossAcc) {
  int wid  = blockIdx.x * 8 + (threadIdx.x >> 5);
  int lane = threadIdx.x & 31;
  int mtile = wid / 6, ntile = wid % 6;
  int row0 = mtile << 4, col0 = ntile << 7;
  v8f acc[8] = {};
  for (int k0 = 0; k0 < D_LAT; k0 += 32) {
    v16h a = load_frag_f16(Zh, D_LAT, row0, k0, lane);
#pragma unroll
    for (int s = 0; s < 8; ++s) {
      v16h b = load_frag_f16(M2t, D_LAT, col0 + s * 16, k0, lane);
      acc[s] = __builtin_amdgcn_wmma_f32_16x16x32_f16(false, a, false, b,
                                                      (short)0, acc[s], false, false);
    }
  }
  int n = lane & 15, mh = lane >> 4;
  float sq = 0.f;
#pragma unroll
  for (int s = 0; s < 8; ++s) {
    int col = col0 + s * 16 + n;
    float bv = b2[col];
#pragma unroll
    for (int v = 0; v < 8; ++v) {
      int row = row0 + v + 8 * mh;
      float r = acc[s][v] + bv;
      float d = r - X[(size_t)row * D_MODEL + col];
      sq += d * d;
    }
  }
#pragma unroll
  for (int off = 16; off > 0; off >>= 1) sq += __shfl_down(sq, off, 32);
  if (lane == 0) atomicAdd(lossAcc, sq);
}

// ---------------------------------------------------------------------------
// Host launcher
// ---------------------------------------------------------------------------
extern "C" void kernel_launch(void* const* d_in, const int* in_sizes, int n_in,
                              void* d_out, int out_size, void* d_ws, size_t ws_size,
                              hipStream_t stream) {
  const float* X      = (const float*)d_in[0];
  const float* enc_kr = (const float*)d_in[1];
  const float* enc_ki = (const float*)d_in[2];
  const float* dec_kr = (const float*)d_in[3];
  const float* dec_ki = (const float*)d_in[4];
  const float* eWw = (const float*)d_in[5];
  const float* eWx = (const float*)d_in[6];
  const float* eWy = (const float*)d_in[7];
  const float* eWz = (const float*)d_in[8];
  const float* enc_b = (const float*)d_in[9];
  const float* dWw = (const float*)d_in[10];
  const float* dWx = (const float*)d_in[11];
  const float* dWy = (const float*)d_in[12];
  const float* dWz = (const float*)d_in[13];
  const float* dec_b = (const float*)d_in[14];

  float* Zout    = (float*)d_out;                       // 32768*384 floats
  float* lossOut = Zout + (size_t)ROWS * D_LAT;         // + 1 float

  char* ws = (char*)d_ws;
  size_t off = 0;
  auto carve = [&](size_t bytes) -> char* {
    char* p = ws + off;
    off = (off + bytes + 255) & ~(size_t)255;
    return p;
  };
  float*     lossAcc = (float*)carve(4);
  float*     k_enc   = (float*)carve((size_t)D_MODEL * 4);
  float*     k_dec   = (float*)carve((size_t)D_MODEL * 4);
  _Float16*  Bt_enc  = (_Float16*)carve((size_t)D_MODEL * D_MODEL * 2);
  _Float16*  Menc    = (_Float16*)carve((size_t)D_LAT * D_MODEL * 2);
  float*     Mdec    = (float*)carve((size_t)D_MODEL * D_LAT * 4);
  _Float16*  M2      = (_Float16*)carve((size_t)D_MODEL * D_LAT * 2);
  float*     b2      = (float*)carve((size_t)D_MODEL * 4);
  _Float16*  H       = (_Float16*)carve((size_t)ROWS * D_MODEL * 2);
  _Float16*  Zh      = (_Float16*)carve((size_t)ROWS * D_LAT * 2);
  (void)ws_size; (void)in_sizes; (void)n_in; (void)out_size;

  // Precompute (all tiny)
  k_zero<<<1, 32, 0, stream>>>(lossAcc);
  k_irfft768<<<1, D_MODEL, 0, stream>>>(enc_kr, enc_ki, k_enc);
  k_irfft768<<<1, D_MODEL, 0, stream>>>(dec_kr, dec_ki, k_dec);
  k_build_Bt<<<(D_MODEL * D_MODEL + 255) / 256, 256, 0, stream>>>(k_enc, Bt_enc);
  k_build_Menc<<<(D_LAT * D_MODEL + 255) / 256, 256, 0, stream>>>(eWw, eWx, eWy, eWz, Menc);
  k_build_Mdec<<<(D_MODEL * D_LAT + 255) / 256, 256, 0, stream>>>(dWw, dWx, dWy, dWz, Mdec);
  k_build_M2<<<(D_MODEL * D_LAT + 255) / 256, 256, 0, stream>>>(Mdec, k_dec, dec_b, M2, b2);

  // Main pipeline: three WMMA GEMMs
  // GEMM1: waves = (32768/16)*(768/128) = 12288 -> 1536 blocks of 8 waves
  k_gemm_filter_gelu<<<1536, 256, 0, stream>>>(X, Bt_enc, H);
  // GEMM2: waves = 2048*3 = 6144 -> 768 blocks
  k_gemm_enc_quat<<<768, 256, 0, stream>>>(H, Menc, enc_b, Zout, Zh);
  // GEMM3 + fused loss: 12288 waves -> 1536 blocks
  k_gemm_dec_loss<<<1536, 256, 0, stream>>>(Zh, M2, b2, X, lossAcc);
  k_final<<<1, 32, 0, stream>>>(lossAcc, lossOut);
}
